// NLinearMemoryEfficient_59571196396198
// MI455X (gfx1250) — compile-verified, matrix operand order checked by reading, not run on hardware
//
#include <hip/hip_runtime.h>

// ---------------------------------------------------------------------------
// NLinear (256 independent 2048x256 @ 256x256 GEMMs + bias) for MI455X/gfx1250
//
// Memory-bound (~1.1 GB min traffic @ 23.3 TB/s => ~47us floor vs ~14us bf16
// WMMA compute). bf16 WMMA w/ f32 accumulation; W pre-converted+transposed to
// bf16 Wt[n][o][k] in workspace (32 MB, L2-resident); x (A) tiles DMA'd into
// LDS by the Tensor Data Mover (double-buffered, TENSORcnt-tracked) with TDM
// row padding to avoid LDS bank conflicts; per-wave fp32->bf16 convert on the
// fragment read. B fragments load straight from L2-resident Wt.
// ---------------------------------------------------------------------------

typedef __attribute__((ext_vector_type(16))) __bf16 v16bf; // WMMA A/B frag (8 VGPRs)
typedef __attribute__((ext_vector_type(8)))  __bf16 v8bf;  // 16-byte chunk
typedef __attribute__((ext_vector_type(8)))  float  v8f;   // WMMA C/D frag (8 VGPRs)
typedef __attribute__((ext_vector_type(4)))  float  v4f;
typedef __attribute__((ext_vector_type(4)))  unsigned int u32x4;
typedef __attribute__((ext_vector_type(8)))  int          i32x8;
typedef __attribute__((ext_vector_type(4)))  int          i32x4;

#define NB 256   // number of slices (n)
#define DK 256   // d_in
#define DO 256   // d_out
#define BB 2048  // batch
#define MT 64    // output rows per workgroup
#define KC 32    // K chunk per WMMA step
#define KSTEPS (DK / KC)

#if defined(__AMDGCN__) && __has_builtin(__builtin_amdgcn_tensor_load_to_lds)
#define USE_TDM 1
#else
#define USE_TDM 0
#endif

// TDM pads 4 DWORDs (16B) after every 32 DWORDs (one 128B tile row):
#define AROW_BYTES 144           // 32*4 + 16 pad
#define ATILE_BYTES (MT * AROW_BYTES)

// ---- pre-pass: W[n][k][o] fp32  ->  Wt[n][o][k] bf16 (transposed) ----------
__global__ void __launch_bounds__(256)
wt_transpose_kernel(const float* __restrict__ W, unsigned short* __restrict__ Wt_) {
  __bf16* Wt = reinterpret_cast<__bf16*>(Wt_);
  const int n = blockIdx.x;   // 0..255
  const int o = threadIdx.x;  // 0..255
  const float* Wn  = W  + (size_t)n * DK * DO;
  __bf16*      Wtn = Wt + ((size_t)n * DO + o) * DK;
  for (int k0 = 0; k0 < DK; k0 += 8) {
    v8bf h;
#pragma unroll
    for (int j = 0; j < 8; ++j)
      h[j] = (__bf16)Wn[(size_t)(k0 + j) * DO + o];  // coalesced across threads
    *reinterpret_cast<v8bf*>(Wtn + k0) = h;          // 16B store, contiguous per thread
  }
}

#if USE_TDM
// Issue one TDM 2D tile load: 32 fp32 x 64 rows from global (row stride
// 65536 elems) into LDS at lds_addr with 16B pad per 128B row.
// D# packing per CDNA5 ISA section 8.3/8.4 (groups 0..3); 6-arg builtin form.
__device__ __forceinline__ void tdm_load_a_tile(unsigned lds_addr,
                                                unsigned long long gaddr) {
  u32x4 g0;
  g0[0] = 1u;                                        // count=1, user descriptor
  g0[1] = lds_addr;                                  // lds_addr [63:32]
  g0[2] = (unsigned)gaddr;                           // global_addr [95:64]
  g0[3] = ((unsigned)(gaddr >> 32) & 0x01FFFFFFu)    // global_addr [120:96]
          | (2u << 30);                              // type=2 ("image")
  i32x8 g1;
  g1[0] = (int)((2u << 16)      // data_size: 4 bytes
                | (1u << 20)    // pad_enable
                | (4u << 22)    // pad_interval: 32 DWORDs (one tile row)
                | (3u << 25));  // pad_amount: 4 DWORDs (16B)
  g1[1] = (int)((unsigned)KC << 16);   // tensor_dim0 = 32 (lo16 @ [63:48])
  g1[2] = (int)((unsigned)MT << 16);   // dim0 hi=0 | tensor_dim1 = 64 (lo16)
  g1[3] = (int)((unsigned)KC << 16);   // dim1 hi=0 | tile_dim0 = 32
  g1[4] = MT;                          // tile_dim1 = 64 | tile_dim2 = 0
  g1[5] = NB * DK;                     // tensor_dim0_stride = 65536 (lo32)
  g1[6] = 0;                           // stride0 hi | stride1 lo
  g1[7] = 0;                           // stride1 hi
  i32x4 g2; g2[0] = 0; g2[1] = 0; g2[2] = 0; g2[3] = 0;
  i32x4 g3; g3[0] = 0; g3[1] = 0; g3[2] = 0; g3[3] = 0;
  i32x8 g4; g4[0] = 0; g4[1] = 0; g4[2] = 0; g4[3] = 0;
            g4[4] = 0; g4[5] = 0; g4[6] = 0; g4[7] = 0;
  __builtin_amdgcn_tensor_load_to_lds(g0, g1, g2, g3, g4, 0);
}
#endif

// ---- main kernel: per (row-tile, slice) computes 64x256 output tile --------
__global__ void __launch_bounds__(256)
nlinear_wmma_kernel(const float* __restrict__ x,
                    const unsigned short* __restrict__ Wt_,
                    const float* __restrict__ bias,
                    float* __restrict__ out) {
  const __bf16* Wt = reinterpret_cast<const __bf16*>(Wt_);

  const int rt = blockIdx.x;       // row tile 0..31
  const int n  = blockIdx.y;       // slice 0..255
  const int m0 = rt * MT;

  const int tid  = threadIdx.x;
  const int wave = tid >> 5;       // 8 waves (wave32)
  const int lane = tid & 31;
  const int l15  = lane & 15;
  const int sel  = lane >> 4;      // K-chunk selector per ISA 16-bit layout
  const int wr   = wave & 3;       // row group (16 rows each)
  const int wcg  = wave >> 2;      // col group (128 cols each)

  // B fragment base: Wt[n][o][sel*8 ...]
  const int obase = wcg * 128;
  const __bf16* wb = Wt + (size_t)n * DO * DK + sel * 8;

  v8f acc[8];
#pragma unroll
  for (int c = 0; c < 8; ++c)
#pragma unroll
    for (int j = 0; j < 8; ++j) acc[c][j] = 0.0f;

#if USE_TDM
  // ---- TDM double-buffered fp32 A tiles in LDS ----
  __shared__ __align__(16) unsigned char As[2][ATILE_BYTES];
  const unsigned lds_a[2] = { (unsigned)(uintptr_t)&As[0][0],
                              (unsigned)(uintptr_t)&As[1][0] };
  const unsigned long long gb =
      (unsigned long long)(uintptr_t)(x + ((size_t)m0 * NB + n) * DK);

  // Per-wave fragment read offsets (fp32 rows, 144B stride, +16B pad/row):
  //   lane reads K chunks {sel*8..+7} and {16+sel*8..+7}  (bytes sel*32, 64+sel*32)
  const unsigned aoff = (unsigned)((wr * 16 + l15) * AROW_BYTES + sel * 32);
  const unsigned char* a_base[2] = { &As[0][aoff], &As[1][aoff] };

  if (wave == 0) tdm_load_a_tile(lds_a[0], gb);  // prime buffer 0 (k0 = 0)

  int buf = 0;
  for (int step = 0; step < KSTEPS; ++step) {
    const int k0 = step * KC;
    __syncthreads();  // readers of the buffer we are about to refill are done
    if (wave == 0) {
      if (step + 1 < KSTEPS) {
        tdm_load_a_tile(lds_a[buf ^ 1], gb + (unsigned long long)(k0 + KC) * 4u);
        __builtin_amdgcn_s_wait_tensorcnt(1);  // current buffer's DMA complete
      } else {
        __builtin_amdgcn_s_wait_tensorcnt(0);  // last tile complete
      }
    }
    __syncthreads();  // current buffer's tile visible to all waves

    // ---- A fragment: 16 fp32 from LDS -> bf16 ----
    v16bf a;
    {
      const float* ap = reinterpret_cast<const float*>(a_base[buf]);
      v4f q0 = *reinterpret_cast<const v4f*>(ap);          // K sel*8 + 0..3
      v4f q1 = *reinterpret_cast<const v4f*>(ap + 4);      // K sel*8 + 4..7
      v4f q2 = *reinterpret_cast<const v4f*>(ap + 16);     // K 16+sel*8 + 0..3
      v4f q3 = *reinterpret_cast<const v4f*>(ap + 20);     // K 16+sel*8 + 4..7
#pragma unroll
      for (int j = 0; j < 4; ++j) {
        a[j]      = (__bf16)q0[j];
        a[4 + j]  = (__bf16)q1[j];
        a[8 + j]  = (__bf16)q2[j];
        a[12 + j] = (__bf16)q3[j];
      }
    }

    // ---- 8 column tiles: B frag from L2-resident Wt, then WMMA ----
#pragma unroll
    for (int c = 0; c < 8; ++c) {
      const int o = obase + c * 16 + l15;
      const __bf16* bp = wb + (size_t)o * DK + k0;
      v8bf lo = *reinterpret_cast<const v8bf*>(bp);
      v8bf hi = *reinterpret_cast<const v8bf*>(bp + 16);
      v16bf bm;
#pragma unroll
      for (int j = 0; j < 8; ++j) { bm[j] = lo[j]; bm[8 + j] = hi[j]; }
      acc[c] = __builtin_amdgcn_wmma_f32_16x16x32_bf16(
          false, a, false, bm, (short)0, acc[c], false, false);
    }
    buf ^= 1;
  }
#else
  // ---- fallback staging path (no TDM builtin): cvt in staging, bf16 LDS ----
  __shared__ __align__(16) __bf16 Asf[MT][40];
  const int sr = tid >> 2;
  const int sk = (tid & 3) * 8;
  const float* xg = x + ((size_t)(m0 + sr) * NB + n) * DK;
  const __bf16* afp = &Asf[wr * 16 + l15][sel * 8];

  for (int k0 = 0; k0 < DK; k0 += KC) {
    __syncthreads();
    v4f f0 = *reinterpret_cast<const v4f*>(xg + k0 + sk);
    v4f f1 = *reinterpret_cast<const v4f*>(xg + k0 + sk + 4);
    if (k0 + KC < DK) __builtin_prefetch(xg + k0 + KC + sk, 0, 0);
    v8bf h;
#pragma unroll
    for (int j = 0; j < 4; ++j) { h[j] = (__bf16)f0[j]; h[4 + j] = (__bf16)f1[j]; }
    *reinterpret_cast<v8bf*>(&Asf[sr][sk]) = h;
    __syncthreads();

    v16bf a;
    {
      v8bf lo = *reinterpret_cast<const v8bf*>(afp);
      v8bf hi = *reinterpret_cast<const v8bf*>(afp + 16);
#pragma unroll
      for (int j = 0; j < 8; ++j) { a[j] = lo[j]; a[8 + j] = hi[j]; }
    }
#pragma unroll
    for (int c = 0; c < 8; ++c) {
      const int o = obase + c * 16 + l15;
      const __bf16* bp = wb + (size_t)o * DK + k0;
      v8bf lo = *reinterpret_cast<const v8bf*>(bp);
      v8bf hi = *reinterpret_cast<const v8bf*>(bp + 16);
      v16bf bm;
#pragma unroll
      for (int j = 0; j < 8; ++j) { bm[j] = lo[j]; bm[8 + j] = hi[j]; }
      acc[c] = __builtin_amdgcn_wmma_f32_16x16x32_bf16(
          false, a, false, bm, (short)0, acc[c], false, false);
    }
  }
#endif

  // ---- epilogue: C layout (VGPR j -> M=j or j+8 by lane half), + bias ----
#pragma unroll
  for (int c = 0; c < 8; ++c) {
    const int o = obase + c * 16 + l15;
    const float bv = bias[(size_t)n * DO + o];
#pragma unroll
    for (int j = 0; j < 8; ++j) {
      const int m = m0 + wr * 16 + sel * 8 + j;
      out[((size_t)m * NB + n) * DO + o] = acc[c][j] + bv;
    }
  }
}

// ---- fallback (workspace too small for Wt): naive fp32 dot product ---------
__global__ void __launch_bounds__(256)
nlinear_fallback_kernel(const float* __restrict__ x, const float* __restrict__ W,
                        const float* __restrict__ bias, float* __restrict__ out) {
  const size_t idx = (size_t)blockIdx.x * 256 + threadIdx.x; // over B*N*D_OUT
  const int o = (int)(idx & 255);
  const size_t bn = idx >> 8;
  const int n = (int)(bn & 255);
  const size_t b = bn >> 8;
  const float* xr = x + (b * NB + n) * DK;
  const float* wc = W + (size_t)n * DK * DO + o;
  float s = bias[(size_t)n * DO + o];
  for (int k = 0; k < DK; ++k) s += xr[k] * wc[(size_t)k * DO];
  out[idx] = s;
}

extern "C" void kernel_launch(void* const* d_in, const int* in_sizes, int n_in,
                              void* d_out, int out_size, void* d_ws, size_t ws_size,
                              hipStream_t stream) {
  const float* x    = (const float*)d_in[0];  // [2048, 256, 256] fp32
  const float* W    = (const float*)d_in[1];  // [256, 256, 256] fp32
  const float* bias = (const float*)d_in[2];  // [256, 256] fp32
  float* out = (float*)d_out;                 // [2048, 256, 256] fp32

  const size_t wt_bytes = (size_t)NB * DO * DK * sizeof(unsigned short); // 33.5 MB
  if (ws_size >= wt_bytes) {
    unsigned short* Wt = (unsigned short*)d_ws;
    wt_transpose_kernel<<<dim3(NB), dim3(256), 0, stream>>>(W, Wt);
    nlinear_wmma_kernel<<<dim3(BB / MT, NB), dim3(256), 0, stream>>>(x, Wt, bias, out);
  } else {
    const size_t total = (size_t)BB * NB * DO;
    nlinear_fallback_kernel<<<dim3((unsigned)(total / 256)), dim3(256), 0, stream>>>(
        x, W, bias, out);
  }
}
